// CATSI_53910429500029
// MI455X (gfx1250) — matrix operand order repaired and checked by reading, not compiled, not run on hardware
//
#include <hip/hip_runtime.h>
#include <math.h>

typedef __attribute__((ext_vector_type(16))) _Float16 v16h;
typedef __attribute__((ext_vector_type(8)))  _Float16 v8h;
typedef __attribute__((ext_vector_type(8)))  float    v8f;

#define Bdim 128
#define Tdim 256
#define Vdim 64
#define BT   32768   // Bdim*Tdim
#define BTV  2097152 // BT*Vdim

__device__ __forceinline__ float sigf(float x) { return 1.0f / (1.0f + expf(-x)); }

__device__ __forceinline__ v8f wmma16(v16h a, v16h b, v8f c) {
    // D = A(16x32 f16) * B(32x16 f16) + C(16x16 f32)
    return __builtin_amdgcn_wmma_f32_16x16x32_f16(false, a, false, b, (short)0, c, false, false);
}

// ---------------------------------------------------------------------------
// Stats: msum, means, stdev, missing; stats vector; nmeans; mmn
// grid=B, block=64 (one thread per v)
// ---------------------------------------------------------------------------
__global__ void k_stats(const float* __restrict__ values, const float* __restrict__ masks,
                        const int* __restrict__ lengths,
                        const float* __restrict__ minv, const float* __restrict__ maxv,
                        float* __restrict__ stats, float* __restrict__ nmeans,
                        float* __restrict__ mmn) {
    int b = blockIdx.x, v = threadIdx.x;
    const float* vp = values + (size_t)b * Tdim * Vdim + v;
    const float* mp = masks  + (size_t)b * Tdim * Vdim + v;
    float s = 0.f, ms = 0.f;
    for (int t = 0; t < Tdim; ++t) { s += vp[t * Vdim]; ms += mp[t * Vdim]; }
    float mean = s / ms;
    float var = 0.f;
    for (int t = 0; t < Tdim; ++t) { float d = vp[t * Vdim] - mean; var += d * d; }
    var /= (ms - 1.0f);
    float sd = sqrtf(var);
    float lf = (float)lengths[b];
    float miss = 1.0f - ms / lf;
    float* st = stats + (size_t)b * 224;
    if (v == 0) st[0] = lf;
    st[1 + v] = mean;
    st[65 + v] = sd;
    st[129 + v] = miss;
    float mn = minv[b * Vdim + v], mx = maxv[b * Vdim + v];
    float d = fmaxf(mx - mn, 1e-8f);
    mmn[b * Vdim + v] = d;
    nmeans[b * Vdim + v] = (mean - mn) / d;
}

// ---------------------------------------------------------------------------
// Context MLP: ctx_mlp = relu(stats@Wc1^T+bc1)@Wc2^T+bc2. grid=B, block=64
// ---------------------------------------------------------------------------
__global__ void k_ctx(const float* __restrict__ stats,
                      const float* __restrict__ Wc1, const float* __restrict__ bc1,
                      const float* __restrict__ Wc2, const float* __restrict__ bc2,
                      float* __restrict__ ctx_mlp) {
    __shared__ float st[193];
    __shared__ float hid[64];
    int b = blockIdx.x, i = threadIdx.x;
    for (int k = i; k < 193; k += 64) st[k] = stats[(size_t)b * 224 + k];
    __syncthreads();
    float acc = bc1[i];
    for (int k = 0; k < 193; ++k) acc += st[k] * Wc1[i * 193 + k];
    hid[i] = fmaxf(acc, 0.f);
    __syncthreads();
    if (i < 32) {
        float a = bc2[i];
        for (int k = 0; k < 64; ++k) a += hid[k] * Wc2[i * 64 + k];
        ctx_mlp[b * 32 + i] = a;
    }
}

// ---------------------------------------------------------------------------
// Forward-fill + decay + completion. grid=B, block=64 (thread per v, loop t)
// Writes x_complete f32, gamma f32, and [x_complete|masks] as f16 rows (BT x 128)
// ---------------------------------------------------------------------------
__global__ void k_complete(const float* __restrict__ values, const float* __restrict__ masks,
                           const float* __restrict__ deltas,
                           const float* __restrict__ minv, const float* __restrict__ mmn,
                           const float* __restrict__ nmeans,
                           const float* __restrict__ W_decay, const float* __restrict__ b_decay,
                           float* __restrict__ xc, float* __restrict__ gam,
                           _Float16* __restrict__ xm) {
    int b = blockIdx.x, v = threadIdx.x;
    float mn = minv[b * Vdim + v];
    float d = mmn[b * Vdim + v];
    float nm = nmeans[b * Vdim + v];
    float wd = W_decay[v * Vdim + v];
    float bd = b_decay[v];
    float carry = 0.f, nvprev = 0.f, mprev = 0.f;
    for (int t = 0; t < Tdim; ++t) {
        size_t o = ((size_t)b * Tdim + t) * Vdim + v;
        float nv = (values[o] - mn) / d;
        float m = masks[o];
        float xp;
        if (t == 0) { carry = nv; xp = nv; }
        else { carry = mprev * nvprev + (1.f - mprev) * carry; xp = carry; }
        float g = expf(-fmaxf(deltas[o] * wd + bd, 0.f));
        float xd = g * xp + (1.f - g) * nm;
        float x = m * nv + (1.f - m) * xd;
        xc[o] = x;
        gam[o] = g;
        size_t ro = ((size_t)b * Tdim + t) * 128;
        xm[ro + v] = (_Float16)x;
        xm[ro + 64 + v] = (_Float16)m;
        nvprev = nv; mprev = m;
    }
}

// ---------------------------------------------------------------------------
// Pack a (N,K)-rowmajor f32 weight (used as x@W^T) into CDNA5 WMMA B-layout f16:
// out[((kc*Nt+nt)*32+lane)*16+e] = W[n][koff+k], k=kc*32+(lane>=16?8:0)+(e&7)+(e>=8?16:0), n=nt*16+(lane&15)
// zero_diag_shift>=0: zero when (n>>shift)==k  (masks W_feat diagonal)
// ---------------------------------------------------------------------------
__global__ void k_pack(const float* __restrict__ W, int ldw, int koff, int K, int N,
                       int Nt, int zero_diag_shift, _Float16* __restrict__ out, int total) {
    int idx = blockIdx.x * blockDim.x + threadIdx.x;
    if (idx >= total) return;
    int e = idx & 15;
    int lane = (idx >> 4) & 31;
    int nt = (idx >> 9) % Nt;
    int kc = idx / (512 * Nt);
    int k = kc * 32 + ((lane & 16) ? 8 : 0) + (e & 7) + ((e & 8) ? 16 : 0);
    int n = nt * 16 + (lane & 15);
    float v = 0.f;
    if (k < K && n < N) {
        v = W[(size_t)n * ldw + koff + k];
        if (zero_diag_shift >= 0 && (n >> zero_diag_shift) == k) v = 0.f;
    }
    out[idx] = (_Float16)v;
}

// ---------------------------------------------------------------------------
// WMMA GEMM, 2 N-tiles per wave (A-frag reuse): C = A(M x K f16 rowmajor)*Bpack + colbias
// grid=(M/16, Nt/2), block=32. Cf or Ch selects f32/f16 output.
// ---------------------------------------------------------------------------
__global__ void k_gemm2(const _Float16* __restrict__ A, int lda, int Kc,
                        const _Float16* __restrict__ Bp, int Nt,
                        const float* __restrict__ bias,
                        float* __restrict__ Cf, _Float16* __restrict__ Ch, int ldc) {
    int mt = blockIdx.x;
    int nt0 = blockIdx.y * 2;
    int lane = threadIdx.x;
    int r = lane & 15;
    int kof = (lane & 16) ? 8 : 0;
    const _Float16* arow = A + (size_t)(mt * 16 + r) * lda;
    v8f acc0 = {}, acc1 = {};
    for (int kc = 0; kc < Kc; ++kc) {
        int kb = kc * 32 + kof;
        v8h lo = *(const v8h*)(arow + kb);
        v8h hi = *(const v8h*)(arow + kb + 16);
        v16h a;
#pragma unroll
        for (int i = 0; i < 8; ++i) { a[i] = lo[i]; a[8 + i] = hi[i]; }
        v16h b0 = *(const v16h*)(Bp + (((size_t)kc * Nt + nt0) * 32 + lane) * 16);
        v16h b1 = *(const v16h*)(Bp + (((size_t)kc * Nt + nt0 + 1) * 32 + lane) * 16);
        acc0 = wmma16(a, b0, acc0);
        acc1 = wmma16(a, b1, acc1);
    }
    int rbase = (lane & 16) ? 8 : 0;
    int col0 = nt0 * 16 + r;
    float bv0 = bias ? bias[col0] : 0.f;
    float bv1 = bias ? bias[col0 + 16] : 0.f;
    if (Ch) {
#pragma unroll
        for (int j = 0; j < 8; ++j) {
            size_t o = (size_t)(mt * 16 + rbase + j) * ldc + col0;
            Ch[o] = (_Float16)(acc0[j] + bv0);
            Ch[o + 16] = (_Float16)(acc1[j] + bv1);
        }
    } else {
#pragma unroll
        for (int j = 0; j < 8; ++j) {
            size_t o = (size_t)(mt * 16 + rbase + j) * ldc + col0;
            Cf[o] = acc0[j] + bv0;
            Cf[o + 16] = acc1[j] + bv1;
        }
    }
}

// ---------------------------------------------------------------------------
// GRU scan over T steps for a 16-batch tile. grid=8, block=64 (2 waves).
// Wave w owns hidden cols j in [16w,16w+16): N-tiles {w, 2+w, 4+w}.
// gi precomputed in G (f32, bias folded). Recurrent gh via WMMA (K=32, N=96).
// ---------------------------------------------------------------------------
__global__ void k_gru(const float* __restrict__ G, const float* __restrict__ b_hh,
                      const _Float16* __restrict__ Pghh, const int* __restrict__ lengths,
                      float* __restrict__ ctx_rnn) {
    __shared__ _Float16 h16[16 * 32];
    __shared__ float h32[16 * 32];
    __shared__ float hn[16 * 32];
    int tid = threadIdx.x;
    int lane = tid & 31;
    int w = tid >> 5;                // 0..1
    int bt = blockIdx.x;
    int c = lane & 15;
    int rbase = (lane & 16) ? 8 : 0;
    int j = c + 16 * w;
    for (int i = tid; i < 512; i += 64) { h16[i] = (_Float16)0.f; h32[i] = 0.f; hn[i] = 0.f; }
    __syncthreads();
    for (int t = 0; t < Tdim; ++t) {
        int r = lane & 15;
        int kb = (lane & 16) ? 8 : 0;
        v8h lo = *(const v8h*)(&h16[r * 32 + kb]);
        v8h hi = *(const v8h*)(&h16[r * 32 + kb + 16]);
        v16h a;
#pragma unroll
        for (int i = 0; i < 8; ++i) { a[i] = lo[i]; a[8 + i] = hi[i]; }
        v8f accr = {}, accz = {}, accn = {};
        {
            v16h br = *(const v16h*)(Pghh + ((size_t)(w) * 32 + lane) * 16);
            v16h bz = *(const v16h*)(Pghh + ((size_t)(2 + w) * 32 + lane) * 16);
            v16h bn = *(const v16h*)(Pghh + ((size_t)(4 + w) * 32 + lane) * 16);
            accr = wmma16(a, br, accr);
            accz = wmma16(a, bz, accz);
            accn = wmma16(a, bn, accn);
        }
        __syncthreads();
        float bhr = b_hh[j], bhz = b_hh[32 + j], bhn = b_hh[64 + j];
#pragma unroll
        for (int j8 = 0; j8 < 8; ++j8) {
            int rr = rbase + j8;
            int b = bt * 16 + rr;
            size_t grow = ((size_t)b * Tdim + t) * 96;
            float p = ((t + 1) <= lengths[b]) ? 1.f : 0.f;
            float gir = G[grow + j];
            float giz = G[grow + 32 + j];
            float gin = G[grow + 64 + j];
            float rg = sigf(gir + accr[j8] + bhr);
            float zg = sigf(giz + accz[j8] + bhz);
            float ng = tanhf(gin + rg * (accn[j8] + bhn));
            float hold = h32[rr * 32 + j];
            float hnew = (1.f - zg) * ng + zg * hold;
            hn[rr * 32 + j] = p * hnew + (1.f - p) * hn[rr * 32 + j];
            h32[rr * 32 + j] = hnew;
            h16[rr * 32 + j] = (_Float16)hnew;
        }
        __syncthreads();
    }
#pragma unroll
    for (int j8 = 0; j8 < 8; ++j8) {
        int rr = rbase + j8;
        int b = bt * 16 + rr;
        ctx_rnn[b * 32 + j] = hn[rr * 32 + j];
    }
}

// ---------------------------------------------------------------------------
// context assembly, hinit/cinit, ctx-part of LSTM gates. grid=B, block=256
// ---------------------------------------------------------------------------
__global__ void k_init(const float* __restrict__ ctx_mlp, const float* __restrict__ ctx_rnn,
                       const float* __restrict__ W_init, const float* __restrict__ b_init,
                       const float* __restrict__ W_ih_f, const float* __restrict__ b_ih_f,
                       const float* __restrict__ b_hh_f,
                       const float* __restrict__ W_ih_b, const float* __restrict__ b_ih_b,
                       const float* __restrict__ b_hh_b,
                       float* __restrict__ context, float* __restrict__ hinit,
                       float* __restrict__ cinit,
                       float* __restrict__ ctxg_f, float* __restrict__ ctxg_b) {
    __shared__ float cx[64];
    int b = blockIdx.x, i = threadIdx.x;
    if (i < 32) cx[i] = ctx_mlp[b * 32 + i];
    else if (i < 64) cx[i] = ctx_rnn[b * 32 + (i - 32)];
    __syncthreads();
    if (i < 64) context[b * 64 + i] = cx[i];
    if (i < 128) {
        float a = b_init[i];
        for (int k = 0; k < 64; ++k) a += cx[k] * W_init[i * 64 + k];
        hinit[b * 128 + i] = a;
        cinit[b * 128 + i] = tanhf(a);
    }
    float af = b_ih_f[i] + b_hh_f[i];
    float ab = b_ih_b[i] + b_hh_b[i];
    for (int k = 0; k < 64; ++k) {
        af += cx[k] * W_ih_f[i * 192 + 128 + k];
        ab += cx[k] * W_ih_b[i * 192 + 128 + k];
    }
    ctxg_f[b * 256 + i] = af;
    ctxg_b[b * 256 + i] = ab;
}

// ---------------------------------------------------------------------------
// LSTM scan (one direction) for a 16-batch tile. grid=8, block=128 (4 waves).
// Wave w owns hidden cols j in [16w,16w+16): gate N-tiles {w, 4+w, 8+w, 12+w}.
// Recurrent h@Whh^T via WMMA (K=64 -> 2 chunks). dir=0 fwd, dir=1 bwd.
// ---------------------------------------------------------------------------
__global__ void k_lstm(const _Float16* __restrict__ Gh, const float* __restrict__ ctxg,
                       const _Float16* __restrict__ Phh,
                       const float* __restrict__ hinit, const float* __restrict__ cinit,
                       float* __restrict__ h_all, int dir) {
    __shared__ _Float16 h16[16 * 64];
    __shared__ float cst[16 * 64];
    int tid = threadIdx.x;
    int lane = tid & 31;
    int w = tid >> 5;                // 0..3
    int bt = blockIdx.x;
    int c = lane & 15;
    int rbase = (lane & 16) ? 8 : 0;
    int j = c + 16 * w;
    int hoff = dir ? 64 : 0;
    int t0 = dir ? (Tdim - 1) : 0;
#pragma unroll
    for (int j8 = 0; j8 < 8; ++j8) {
        int rr = rbase + j8;
        int b = bt * 16 + rr;
        float hv = hinit[b * 128 + hoff + j];
        h16[rr * 64 + j] = (_Float16)hv;
        cst[rr * 64 + j] = cinit[b * 128 + hoff + j];
        h_all[(size_t)t0 * (Bdim * 64) + b * 64 + j] = hv;
    }
    __syncthreads();
    for (int s = 0; s < Tdim - 1; ++s) {
        int t_in  = dir ? (Tdim - 1 - s) : s;
        int t_out = dir ? (Tdim - 2 - s) : (s + 1);
        int r = lane & 15;
        int kof = (lane & 16) ? 8 : 0;
        v16h afr[2];
#pragma unroll
        for (int kc = 0; kc < 2; ++kc) {
            int kb = kc * 32 + kof;
            v8h lo = *(const v8h*)(&h16[r * 64 + kb]);
            v8h hi = *(const v8h*)(&h16[r * 64 + kb + 16]);
#pragma unroll
            for (int i = 0; i < 8; ++i) { afr[kc][i] = lo[i]; afr[kc][8 + i] = hi[i]; }
        }
        v8f acc[4];  // i, f, g, o gate tiles for this wave's j-range
#pragma unroll
        for (int g4 = 0; g4 < 4; ++g4) {
            v8f z = {};
            acc[g4] = z;
            int nt = g4 * 4 + w;
#pragma unroll
            for (int kc = 0; kc < 2; ++kc) {
                v16h bfrag = *(const v16h*)(Phh + (((size_t)kc * 16 + nt) * 32 + lane) * 16);
                acc[g4] = wmma16(afr[kc], bfrag, acc[g4]);
            }
        }
        __syncthreads();
#pragma unroll
        for (int j8 = 0; j8 < 8; ++j8) {
            int rr = rbase + j8;
            int b = bt * 16 + rr;
            size_t grow = ((size_t)b * Tdim + t_in) * 256;
            size_t crow = (size_t)b * 256;
            float gi = (float)Gh[grow + j]       + ctxg[crow + j]       + acc[0][j8];
            float gf = (float)Gh[grow + 64 + j]  + ctxg[crow + 64 + j]  + acc[1][j8];
            float gg = (float)Gh[grow + 128 + j] + ctxg[crow + 128 + j] + acc[2][j8];
            float go = (float)Gh[grow + 192 + j] + ctxg[crow + 192 + j] + acc[3][j8];
            float cc = sigf(gf) * cst[rr * 64 + j] + sigf(gi) * tanhf(gg);
            float hh = sigf(go) * tanhf(cc);
            cst[rr * 64 + j] = cc;
            h16[rr * 64 + j] = (_Float16)hh;
            h_all[(size_t)t_out * (Bdim * 64) + b * 64 + j] = hh;
        }
        __syncthreads();
    }
}

// ---------------------------------------------------------------------------
// Fused feature MLP: hidden=einsum+b_feat, relu, @Wnl1^T, relu, dot Wnl2.
// grid=BT/16, block=32 (one wave per 16 (b,t) rows).
// ---------------------------------------------------------------------------
__global__ void k_feat(const _Float16* __restrict__ xm, const _Float16* __restrict__ Pfeat,
                       const _Float16* __restrict__ Pnl1,
                       const float* __restrict__ b_feat, const float* __restrict__ b_nl1,
                       const float* __restrict__ W_nl2, const float* __restrict__ b_nl2,
                       float* __restrict__ feat) {
    __shared__ _Float16 hb[16 * 32];
    int lane = threadIdx.x;
    int mt = blockIdx.x;
    int c = lane & 15;
    int rbase = (lane & 16) ? 8 : 0;
    int r = lane & 15;
    int kof = (lane & 16) ? 8 : 0;
    const _Float16* arow = xm + (size_t)(mt * 16 + r) * 128;  // x_complete = cols 0..63
    v16h a0, a1;
    {
        v8h lo = *(const v8h*)(arow + kof);
        v8h hi = *(const v8h*)(arow + kof + 16);
        v8h lo2 = *(const v8h*)(arow + 32 + kof);
        v8h hi2 = *(const v8h*)(arow + 32 + kof + 16);
#pragma unroll
        for (int i = 0; i < 8; ++i) {
            a0[i] = lo[i]; a0[8 + i] = hi[i];
            a1[i] = lo2[i]; a1[8 + i] = hi2[i];
        }
    }
    float w2lo = W_nl2[c], w2hi = W_nl2[16 + c];
    float bn1lo = b_nl1[c], bn1hi = b_nl1[16 + c];
    float bn2 = b_nl2[0];
    for (int g = 0; g < 64; ++g) {
        v8f acc0 = {}, acc1 = {};
#pragma unroll
        for (int kc = 0; kc < 2; ++kc) {
            v16h a = kc ? a1 : a0;
            v16h b0 = *(const v16h*)(Pfeat + (((size_t)kc * 128 + 2 * g) * 32 + lane) * 16);
            v16h b1 = *(const v16h*)(Pfeat + (((size_t)kc * 128 + 2 * g + 1) * 32 + lane) * 16);
            acc0 = wmma16(a, b0, acc0);
            acc1 = wmma16(a, b1, acc1);
        }
        float bf0 = b_feat[g * 32 + c], bf1 = b_feat[g * 32 + 16 + c];
#pragma unroll
        for (int j8 = 0; j8 < 8; ++j8) {
            hb[(rbase + j8) * 32 + c]      = (_Float16)fmaxf(acc0[j8] + bf0, 0.f);
            hb[(rbase + j8) * 32 + 16 + c] = (_Float16)fmaxf(acc1[j8] + bf1, 0.f);
        }
        __syncthreads();
        v16h az;
        {
            v8h lo = *(const v8h*)(&hb[r * 32 + kof]);
            v8h hi = *(const v8h*)(&hb[r * 32 + kof + 16]);
#pragma unroll
            for (int i = 0; i < 8; ++i) { az[i] = lo[i]; az[8 + i] = hi[i]; }
        }
        v8f z0 = {}, z1 = {};
        {
            v16h b0 = *(const v16h*)(Pnl1 + ((size_t)0 * 32 + lane) * 16);
            v16h b1 = *(const v16h*)(Pnl1 + ((size_t)1 * 32 + lane) * 16);
            z0 = wmma16(az, b0, z0);
            z1 = wmma16(az, b1, z1);
        }
        float p[8];
#pragma unroll
        for (int j8 = 0; j8 < 8; ++j8)
            p[j8] = fmaxf(z0[j8] + bn1lo, 0.f) * w2lo + fmaxf(z1[j8] + bn1hi, 0.f) * w2hi;
#pragma unroll
        for (int m = 1; m < 16; m <<= 1) {
#pragma unroll
            for (int j8 = 0; j8 < 8; ++j8) p[j8] += __shfl_xor(p[j8], m, 32);
        }
        if ((lane & 15) == 0) {
#pragma unroll
            for (int j8 = 0; j8 < 8; ++j8)
                feat[(size_t)(mt * 16 + rbase + j8) * Vdim + g] = p[j8] + bn2;
        }
        __syncthreads();
    }
}

// ---------------------------------------------------------------------------
// Final fusion: rnn_imp, beta, final; denormalize; write (3,B,T,V).
// grid=B*T, block=64.
// ---------------------------------------------------------------------------
__global__ void k_final(const float* __restrict__ hf, const float* __restrict__ hbk,
                        const float* __restrict__ feat,
                        const float* __restrict__ gam, const float* __restrict__ masks,
                        const float* __restrict__ W_ri, const float* __restrict__ b_ri,
                        const float* __restrict__ W_fu, const float* __restrict__ b_fu,
                        const float* __restrict__ mmn, const float* __restrict__ minv,
                        float* __restrict__ out) {
    __shared__ float hc[128];
    __shared__ float gm[64];
    __shared__ float ms[64];
    int row = blockIdx.x;
    int b = row >> 8, t = row & 255;
    int v = threadIdx.x;
    hc[v]      = hf[(size_t)t * (Bdim * 64) + b * 64 + v];
    hc[64 + v] = hbk[(size_t)t * (Bdim * 64) + b * 64 + v];
    size_t o = (size_t)row * Vdim + v;
    gm[v] = gam[o];
    ms[v] = masks[o];
    __syncthreads();
    float rnn = b_ri[v];
    for (int j = 0; j < 128; ++j) rnn += hc[j] * W_ri[v * 128 + j];
    float ba = b_fu[v];
    for (int k = 0; k < 64; ++k) ba += gm[k] * W_fu[v * 128 + k] + ms[k] * W_fu[v * 128 + 64 + k];
    float beta = sigf(ba);
    float ft = feat[o];
    float fin = beta * ft + (1.f - beta) * rnn;
    float dd = mmn[b * Vdim + v], mn = minv[b * Vdim + v];
    out[o]            = rnn * dd + mn;
    out[BTV + o]      = ft * dd + mn;
    out[2 * BTV + o]  = fin * dd + mn;
}

// ---------------------------------------------------------------------------
extern "C" void kernel_launch(void* const* d_in, const int* in_sizes, int n_in,
                              void* d_out, int out_size, void* d_ws, size_t ws_size,
                              hipStream_t stream) {
    (void)in_sizes; (void)n_in; (void)out_size; (void)ws_size;
    const int*   lengths = (const int*)d_in[0];
    const float* values  = (const float*)d_in[1];
    const float* masks   = (const float*)d_in[2];
    const float* deltas  = (const float*)d_in[3];
    const float* minv    = (const float*)d_in[4];
    const float* maxv    = (const float*)d_in[5];
    const float* W_decay = (const float*)d_in[6];
    const float* b_decay = (const float*)d_in[7];
    const float* W_feat  = (const float*)d_in[8];
    const float* b_feat  = (const float*)d_in[9];
    const float* W_nl1   = (const float*)d_in[10];
    const float* b_nl1   = (const float*)d_in[11];
    const float* W_nl2   = (const float*)d_in[12];
    const float* b_nl2   = (const float*)d_in[13];
    const float* Wc1     = (const float*)d_in[14];
    const float* bc1     = (const float*)d_in[15];
    const float* Wc2     = (const float*)d_in[16];
    const float* bc2     = (const float*)d_in[17];
    const float* W_ih_g  = (const float*)d_in[18];
    const float* W_hh_g  = (const float*)d_in[19];
    const float* b_ih_g  = (const float*)d_in[20];
    const float* b_hh_g  = (const float*)d_in[21];
    const float* W_init  = (const float*)d_in[22];
    const float* b_init  = (const float*)d_in[23];
    const float* W_ih_f  = (const float*)d_in[24];
    const float* W_hh_f  = (const float*)d_in[25];
    const float* b_ih_f  = (const float*)d_in[26];
    const float* b_hh_f  = (const float*)d_in[27];
    const float* W_ih_b  = (const float*)d_in[28];
    const float* W_hh_b  = (const float*)d_in[29];
    const float* b_ih_b  = (const float*)d_in[30];
    const float* b_hh_b  = (const float*)d_in[31];
    const float* W_ri    = (const float*)d_in[32];
    const float* b_ri    = (const float*)d_in[33];
    const float* W_fu    = (const float*)d_in[34];
    const float* b_fu    = (const float*)d_in[35];
    float* out = (float*)d_out;

    char* ws = (char*)d_ws;
    size_t off = 0;
    auto alloc = [&](size_t bytes) -> char* {
        char* p = ws + off;
        off += (bytes + 255) & ~(size_t)255;
        return p;
    };
    float*    stats   = (float*)alloc((size_t)Bdim * 224 * 4);
    float*    nmeans  = (float*)alloc((size_t)Bdim * 64 * 4);
    float*    mmn     = (float*)alloc((size_t)Bdim * 64 * 4);
    float*    xc      = (float*)alloc((size_t)BTV * 4);
    float*    gam     = (float*)alloc((size_t)BTV * 4);
    _Float16* xm      = (_Float16*)alloc((size_t)BT * 128 * 2);
    float*    ctx_mlp = (float*)alloc((size_t)Bdim * 32 * 4);
    float*    ctx_rnn = (float*)alloc((size_t)Bdim * 32 * 4);
    float*    context = (float*)alloc((size_t)Bdim * 64 * 4);
    float*    hinit   = (float*)alloc((size_t)Bdim * 128 * 4);
    float*    cinit   = (float*)alloc((size_t)Bdim * 128 * 4);
    float*    ctxg_f  = (float*)alloc((size_t)Bdim * 256 * 4);
    float*    ctxg_b  = (float*)alloc((size_t)Bdim * 256 * 4);
    float*    G_gru   = (float*)alloc((size_t)BT * 96 * 4);
    _Float16* G_lf    = (_Float16*)alloc((size_t)BT * 256 * 2);
    _Float16* G_lb    = (_Float16*)alloc((size_t)BT * 256 * 2);
    float*    hf_all  = (float*)alloc((size_t)Tdim * Bdim * 64 * 4);
    float*    hb_all  = (float*)alloc((size_t)Tdim * Bdim * 64 * 4);
    float*    featbuf = (float*)alloc((size_t)BTV * 4);
    _Float16* Pgih  = (_Float16*)alloc((size_t)4 * 6 * 512 * 2);
    _Float16* Pfih  = (_Float16*)alloc((size_t)4 * 16 * 512 * 2);
    _Float16* Pbih  = (_Float16*)alloc((size_t)4 * 16 * 512 * 2);
    _Float16* Pfhh  = (_Float16*)alloc((size_t)2 * 16 * 512 * 2);
    _Float16* Pbhh  = (_Float16*)alloc((size_t)2 * 16 * 512 * 2);
    _Float16* Pghh  = (_Float16*)alloc((size_t)1 * 6 * 512 * 2);
    _Float16* Pfeat = (_Float16*)alloc((size_t)2 * 128 * 512 * 2);
    _Float16* Pnl1  = (_Float16*)alloc((size_t)1 * 2 * 512 * 2);

    // 1. stats + normalization constants
    k_stats<<<Bdim, 64, 0, stream>>>(values, masks, lengths, minv, maxv, stats, nmeans, mmn);
    // 2. context MLP
    k_ctx<<<Bdim, 64, 0, stream>>>(stats, Wc1, bc1, Wc2, bc2, ctx_mlp);
    // 3. forward-fill + decay completion
    k_complete<<<Bdim, 64, 0, stream>>>(values, masks, deltas, minv, mmn, nmeans,
                                        W_decay, b_decay, xc, gam, xm);
    // 4. weight packs (f16, WMMA B-layout)
    auto pack = [&](const float* W, int ldw, int koff, int K, int N, int Nt, int Kc,
                    int diag, _Float16* dst) {
        int total = Kc * Nt * 512;
        k_pack<<<(total + 255) / 256, 256, 0, stream>>>(W, ldw, koff, K, N, Nt, diag, dst, total);
    };
    pack(W_ih_g, 128, 0, 128,   96,   6, 4, -1, Pgih);
    pack(W_ih_f, 192, 0, 128,  256,  16, 4, -1, Pfih);
    pack(W_ih_b, 192, 0, 128,  256,  16, 4, -1, Pbih);
    pack(W_hh_f,  64, 0,  64,  256,  16, 2, -1, Pfhh);
    pack(W_hh_b,  64, 0,  64,  256,  16, 2, -1, Pbhh);
    pack(W_hh_g,  32, 0,  32,   96,   6, 1, -1, Pghh);
    pack(W_feat,  64, 0,  64, 2048, 128, 2,  5, Pfeat);   // zero where g==v
    pack(W_nl1,   32, 0,  32,   32,   2, 1, -1, Pnl1);
    // 5. big input-side gate GEMMs (WMMA, 2 N-tiles per wave)
    k_gemm2<<<dim3(BT / 16, 3), 32, 0, stream>>>(xm, 128, 4, Pgih, 6, b_ih_g,
                                                 G_gru, (_Float16*)nullptr, 96);
    k_gemm2<<<dim3(BT / 16, 8), 32, 0, stream>>>(xm, 128, 4, Pfih, 16, (const float*)nullptr,
                                                 (float*)nullptr, G_lf, 256);
    k_gemm2<<<dim3(BT / 16, 8), 32, 0, stream>>>(xm, 128, 4, Pbih, 16, (const float*)nullptr,
                                                 (float*)nullptr, G_lb, 256);
    // 6. GRU scan (recurrent WMMA, 2 waves/tile)
    k_gru<<<8, 64, 0, stream>>>(G_gru, b_hh_g, Pghh, lengths, ctx_rnn);
    // 7. context assembly, LSTM init states + ctx gate contributions
    k_init<<<Bdim, 256, 0, stream>>>(ctx_mlp, ctx_rnn, W_init, b_init,
                                     W_ih_f, b_ih_f, b_hh_f, W_ih_b, b_ih_b, b_hh_b,
                                     context, hinit, cinit, ctxg_f, ctxg_b);
    // 8. bidirectional LSTM scans (recurrent WMMA, 4 waves/tile)
    k_lstm<<<8, 128, 0, stream>>>(G_lf, ctxg_f, Pfhh, hinit, cinit, hf_all, 0);
    k_lstm<<<8, 128, 0, stream>>>(G_lb, ctxg_b, Pbhh, hinit, cinit, hb_all, 1);
    // 9. fused feature MLP (WMMA)
    k_feat<<<BT / 16, 32, 0, stream>>>(xm, Pfeat, Pnl1, b_feat, b_nl1, W_nl2, b_nl2, featbuf);
    // 10. fusion + denormalize
    k_final<<<BT, 64, 0, stream>>>(hf_all, hb_all, featbuf, gam, masks,
                                   W_ri, b_ri, W_fu, b_fu, mmn, minv, out);
}
